// SlidingWindowAttentionLayer_53068615909675
// MI455X (gfx1250) — matrix operand order repaired and checked by reading, compiled, not run on hardware
//
#include <hip/hip_runtime.h>
#include <cstdint>
#include <cstddef>

// ---------------------------------------------------------------------------
// CDNA5 (gfx1250) sliding-window GQA attention layer, bf16 WMMA pipeline.
// wave32; V_WMMA_F32_16X16X32_BF16 for all matrix math, fp32 accumulate.
// V staging via Tensor Data Mover (tensor_load_to_lds + s_wait_tensorcnt).
// ---------------------------------------------------------------------------

typedef __attribute__((ext_vector_type(16))) __bf16 v16bf;
typedef __attribute__((ext_vector_type(8)))  __bf16 v8bf;
typedef __attribute__((ext_vector_type(8)))  float  v8f;
typedef __attribute__((ext_vector_type(4)))  unsigned int u32x4;
typedef __attribute__((ext_vector_type(8)))  unsigned int u32x8;
typedef __attribute__((address_space(3)))    void* lds_void_ptr;

#define D_MODEL 2048
#define N_HEADS 16
#define N_KV    4
#define HD      128
#define WINDOW  512
#define SEQ     2048
#define BATCH   2
#define MROWS   (BATCH*SEQ)   // 4096
#define AWAVES  4             // attention waves per block

__device__ __forceinline__ v16bf pack16(v8bf lo, v8bf hi) {
  return __builtin_shufflevector(lo, hi, 0,1,2,3,4,5,6,7,8,9,10,11,12,13,14,15);
}

__device__ __forceinline__ v8f wmma_bf16(v16bf a, v16bf b, v8f c) {
  return __builtin_amdgcn_wmma_f32_16x16x32_bf16(false, a, false, b,
                                                 (short)0, c, false, false);
}

// 50 * tanh(x/50), overflow-safe
__device__ __forceinline__ float softcap50(float x) {
  float y  = x * 0.02f;
  float t  = __expf(-2.0f * fabsf(y));
  float th = (1.0f - t) / (1.0f + t);
  return 50.0f * copysignf(th, y);
}

// ---------------------------------------------------------------------------
__global__ void cvt_f32_bf16(const float* __restrict__ src,
                             __bf16* __restrict__ dst, int n) {
  int i = blockIdx.x * blockDim.x + threadIdx.x;
  if (i < n) dst[i] = (__bf16)src[i];
}

// ---------------------------------------------------------------------------
// C[M,N] = A[M,K] (bf16, row-major) x W[N,K]^T (bf16, row-major)
// one wave -> 16x64 tile, K stepped by 32, software-pipelined fragments.
__global__ void gemm_bf16_wmma(const __bf16* __restrict__ A,
                               const __bf16* __restrict__ W,
                               void* __restrict__ Cout,
                               int M, int N, int K, int out_f32) {
  int wave  = blockIdx.x * (blockDim.x >> 5) + (threadIdx.x >> 5);
  int Nt    = N >> 6;
  int total = (M >> 4) * Nt;
  if (wave >= total) return;                 // wave-uniform
  int mt = wave / Nt, nt = wave - mt * Nt;
  int m0 = mt << 4,  n0 = nt << 6;
  int lane = threadIdx.x & 31;
  int r = lane & 15, hh = lane >> 4;         // hh = lane-half select

  const __bf16* arow = A + (size_t)(m0 + r) * K;
  const __bf16* wrow[4];
  #pragma unroll
  for (int t = 0; t < 4; ++t) wrow[t] = W + (size_t)(n0 + t * 16 + r) * K;

  v8f acc[4];
  v8f z = {0.f,0.f,0.f,0.f,0.f,0.f,0.f,0.f};
  #pragma unroll
  for (int t = 0; t < 4; ++t) acc[t] = z;

  // prologue: fragments for k0 = 0
  v16bf af = *(const v16bf*)(arow + hh * 16);
  v16bf bfr[4];
  #pragma unroll
  for (int t = 0; t < 4; ++t) bfr[t] = *(const v16bf*)(wrow[t] + hh * 16);

  for (int k0 = 0; k0 < K; k0 += 32) {
    v16bf afc = af;
    v16bf b0 = bfr[0], b1 = bfr[1], b2 = bfr[2], b3 = bfr[3];
    int kn = k0 + 32;
    if (kn < K) {                         // preload next iteration's fragments
      af = *(const v16bf*)(arow + kn + hh * 16);
      #pragma unroll
      for (int t = 0; t < 4; ++t)
        bfr[t] = *(const v16bf*)(wrow[t] + kn + hh * 16);
      __builtin_prefetch(arow + kn + 128, 0, 3);   // near-cache prefetch
    }
    acc[0] = wmma_bf16(afc, b0, acc[0]);
    acc[1] = wmma_bf16(afc, b1, acc[1]);
    acc[2] = wmma_bf16(afc, b2, acc[2]);
    acc[3] = wmma_bf16(afc, b3, acc[3]);
  }

  // D layout: lanes<16 -> M = p, lanes>=16 -> M = p+8 ; N = lane&15
  if (out_f32) {
    float* C = (float*)Cout;
    #pragma unroll
    for (int t = 0; t < 4; ++t)
      #pragma unroll
      for (int p = 0; p < 8; ++p)
        C[(size_t)(m0 + p + 8 * hh) * N + n0 + t * 16 + r] = acc[t][p];
  } else {
    __bf16* C = (__bf16*)Cout;
    #pragma unroll
    for (int t = 0; t < 4; ++t)
      #pragma unroll
      for (int p = 0; p < 8; ++p)
        C[(size_t)(m0 + p + 8 * hh) * N + n0 + t * 16 + r] = (__bf16)acc[t][p];
  }
}

// ---------------------------------------------------------------------------
// In-place RoPE on bf16 [B,S,heads,128]; one thread per (row, pair d<64).
__global__ void rope_kernel(__bf16* __restrict__ X, int heads) {
  int tid   = blockIdx.x * blockDim.x + threadIdx.x;
  int total = BATCH * SEQ * heads * 64;
  if (tid >= total) return;
  int d = tid & 63;
  int h = (tid >> 6) % heads;
  int s = (tid / (64 * heads)) % SEQ;
  int b = tid / (64 * heads * SEQ);
  size_t base = (((size_t)(b * SEQ + s)) * heads + h) * HD;
  float inv = __powf(10000.0f, -(float)d * (1.0f / 64.0f));
  float ang = (float)s * inv;
  float sn, cs;
  __sincosf(ang, &sn, &cs);
  float x0 = (float)X[base + d];
  float x1 = (float)X[base + d + 64];
  X[base + d]      = (__bf16)(x0 * cs - x1 * sn);
  X[base + d + 64] = (__bf16)(x1 * cs + x0 * sn);
}

// ---------------------------------------------------------------------------
// Flash-style sliding-window attention; one wave per 16-query tile per (b,h).
// V block (32 keys x 128 dims) staged to LDS by the Tensor Data Mover,
// overlapped with the Q.K^T + softmax work; fragments then read from LDS.
__global__ void swa_attn_wmma(const __bf16* __restrict__ Q,
                              const __bf16* __restrict__ Kc,
                              const __bf16* __restrict__ V,
                              __bf16* __restrict__ O) {
  __shared__ __align__(32)  __bf16 plds[AWAVES][16 * 32];   // P transpose tile
  __shared__ __align__(256) __bf16 vlds[AWAVES][32 * HD];   // TDM V tile

  int w    = threadIdx.x >> 5;
  int wave = blockIdx.x * AWAVES + w;
  int qt = wave & 127;             // S/16 = 128 query tiles
  int h  = (wave >> 7) & 15;
  int b  = wave >> 11;
  int q0 = qt << 4;
  int lane = threadIdx.x & 31;
  int r = lane & 15, hh = lane >> 4;
  int kvh = h >> 2;                // groups = 4
  const float scale = 0.08838834764831845f;  // 1/sqrt(128)

  __bf16* pl = &plds[w][0];
  __bf16* vl = &vlds[w][0];
  unsigned vlds_off = (unsigned)(unsigned long long)(lds_void_ptr)vl;

  // Resident Q fragments (A-layout), K-dim = 128 -> 4 chunks of 32
  const __bf16* qrow = Q + ((((size_t)(b * SEQ + q0 + r)) * N_HEADS + h) * HD);
  v16bf qf[4];
  #pragma unroll
  for (int c = 0; c < 4; ++c) {
    v8bf lo = *(const v8bf*)(qrow + c * 32 + hh * 16);
    v8bf hi = *(const v8bf*)(qrow + c * 32 + hh * 16 + 8);
    qf[c] = pack16(lo, hi);
  }

  v8f z = {0.f,0.f,0.f,0.f,0.f,0.f,0.f,0.f};
  v8f o[8];
  #pragma unroll
  for (int t = 0; t < 8; ++t) o[t] = z;
  float mst[8], lst[8];
  #pragma unroll
  for (int p = 0; p < 8; ++p) { mst[p] = -1e30f; lst[p] = 0.0f; }

  int wstart = q0 - (WINDOW - 1); if (wstart < 0) wstart = 0;
  int kb0 = wstart & ~31;

  for (int kb = kb0; kb <= q0 + 15; kb += 32) {
    // ---- TDM: async DMA of V[kb..kb+31][0..127] -> vlds -----------------
    // D# group0: count=1, lds_addr, 57-bit global_addr, type=2
    unsigned long long gaddr = (unsigned long long)(uintptr_t)
        (V + (((size_t)(b * SEQ + kb)) * N_KV + kvh) * HD);
    u32x4 g0;
    g0.x = __builtin_amdgcn_readfirstlane(1u);
    g0.y = __builtin_amdgcn_readfirstlane(vlds_off);
    g0.z = __builtin_amdgcn_readfirstlane((unsigned)gaddr);
    g0.w = __builtin_amdgcn_readfirstlane(
              (((unsigned)(gaddr >> 32)) & 0x01FFFFFFu) | 0x80000000u);
    // D# group1: data_size=2B, tensor_dim0=128, tensor_dim1=32,
    //            tile_dim0=128, tile_dim1=32, dim0_stride = N_KV*HD = 512
    u32x8 g1;
    g1[0] = 0x00010000u;          // workgroup_mask=0, data_size=1 (2 bytes)
    g1[1] = ((unsigned)HD) << 16; // tensor_dim0[15:0] = 128
    g1[2] = 32u << 16;            // tensor_dim1[15:0] = 32
    g1[3] = ((unsigned)HD) << 16; // tile_dim0 = 128
    g1[4] = 32u;                  // tile_dim1 = 32, tile_dim2 = 0
    g1[5] = (unsigned)(N_KV * HD);// tensor_dim0_stride = 512 elements
    g1[6] = 0u;
    g1[7] = 0u;
    asm volatile("tensor_load_to_lds %0, %1" :: "s"(g0), "s"(g1) : "memory");

    // ---- S = Q K^T over two 16-key column tiles (overlaps the TDM) ------
    v8f s0 = z, s1 = z;
    const __bf16* krow0 = Kc + ((((size_t)(b * SEQ + kb + r)) * N_KV + kvh) * HD);
    const __bf16* krow1 = krow0 + (size_t)16 * N_KV * HD;
    #pragma unroll
    for (int c = 0; c < 4; ++c) {
      v16bf bf0 = *(const v16bf*)(krow0 + c * 32 + hh * 16);
      s0 = wmma_bf16(qf[c], bf0, s0);
      v16bf bf1 = *(const v16bf*)(krow1 + c * 32 + hh * 16);
      s1 = wmma_bf16(qf[c], bf1, s1);
    }

    // ---- softcap + mask + online softmax (C-layout rows) ----------------
    int key0 = kb + r, key1 = kb + 16 + r;
    #pragma unroll
    for (int p = 0; p < 8; ++p) {
      int rowq = q0 + p + 8 * hh;
      float x0 = softcap50(s0[p] * scale);
      float x1 = softcap50(s1[p] * scale);
      bool a0 = (key0 <= rowq) && (key0 > rowq - WINDOW);
      bool a1 = (key1 <= rowq) && (key1 > rowq - WINDOW);
      x0 = a0 ? x0 : -1e30f;
      x1 = a1 ? x1 : -1e30f;
      float mx = fmaxf(x0, x1);
      #pragma unroll
      for (int d = 1; d < 16; d <<= 1) mx = fmaxf(mx, __shfl_xor(mx, d, 32));
      float mnew = fmaxf(mst[p], mx);
      float corr = __expf(mst[p] - mnew);
      float p0 = (x0 > -1e29f) ? __expf(x0 - mnew) : 0.0f;
      float p1 = (x1 > -1e29f) ? __expf(x1 - mnew) : 0.0f;
      float ls = p0 + p1;
      #pragma unroll
      for (int d = 1; d < 16; d <<= 1) ls += __shfl_xor(ls, d, 32);
      lst[p] = lst[p] * corr + ls;
      mst[p] = mnew;
      #pragma unroll
      for (int t = 0; t < 8; ++t) o[t][p] *= corr;
      // C-layout -> LDS row-major [16 x 32] for the A-fragment re-read
      pl[(p + 8 * hh) * 32 + r]      = (__bf16)p0;
      pl[(p + 8 * hh) * 32 + 16 + r] = (__bf16)p1;
    }

    asm volatile("s_wait_dscnt 0" ::: "memory");  // P transpose fence (intra-wave)

    // ---- P as A-fragment from LDS ---------------------------------------
    v8bf plo = *(const v8bf*)(pl + r * 32 + hh * 16);
    v8bf phi = *(const v8bf*)(pl + r * 32 + hh * 16 + 8);
    v16bf pf = pack16(plo, phi);

    // ---- wait for the TDM tile, V as B-fragment from LDS; O += P V ------
    __builtin_amdgcn_s_wait_tensorcnt(0);
    #pragma unroll
    for (int t = 0; t < 8; ++t) {
      v16bf vf;
      #pragma unroll
      for (int i = 0; i < 8; ++i) {
        vf[2 * i]     = vl[(hh * 16 + 2 * i)     * HD + t * 16 + r];
        vf[2 * i + 1] = vl[(hh * 16 + 2 * i + 1) * HD + t * 16 + r];
      }
      o[t] = wmma_bf16(pf, vf, o[t]);
    }
  }

  // ---- normalize and store bf16 [B,S,H,128] -----------------------------
  float inv[8];
  #pragma unroll
  for (int p = 0; p < 8; ++p) inv[p] = 1.0f / lst[p];
  #pragma unroll
  for (int t = 0; t < 8; ++t)
    #pragma unroll
    for (int p = 0; p < 8; ++p) {
      size_t idx = ((((size_t)(b * SEQ + q0 + p + 8 * hh)) * N_HEADS + h) * HD)
                 + t * 16 + r;
      O[idx] = (__bf16)(o[t][p] * inv[p]);
    }
}

// ---------------------------------------------------------------------------
extern "C" void kernel_launch(void* const* d_in, const int* in_sizes, int n_in,
                              void* d_out, int out_size, void* d_ws, size_t ws_size,
                              hipStream_t stream) {
  (void)in_sizes; (void)n_in; (void)out_size; (void)ws_size;
  const float* hs = (const float*)d_in[0];
  const float* wq = (const float*)d_in[1];
  const float* wk = (const float*)d_in[2];
  const float* wv = (const float*)d_in[3];
  const float* wo = (const float*)d_in[4];

  char* ws = (char*)d_ws;
  size_t off = 0;
  auto alloc = [&](size_t bytes) -> void* {
    void* p = ws + off;
    off += (bytes + 255) & ~(size_t)255;
    return p;
  };
  __bf16* hs_b = (__bf16*)alloc((size_t)MROWS * D_MODEL * 2);
  __bf16* wq_b = (__bf16*)alloc((size_t)2048 * 2048 * 2);
  __bf16* wk_b = (__bf16*)alloc((size_t)512 * 2048 * 2);
  __bf16* wv_b = (__bf16*)alloc((size_t)512 * 2048 * 2);
  __bf16* wo_b = (__bf16*)alloc((size_t)2048 * 2048 * 2);
  __bf16* q_b  = (__bf16*)alloc((size_t)MROWS * 2048 * 2);
  __bf16* k_b  = (__bf16*)alloc((size_t)MROWS * 512 * 2);
  __bf16* v_b  = (__bf16*)alloc((size_t)MROWS * 512 * 2);
  __bf16* at_b = (__bf16*)alloc((size_t)MROWS * 2048 * 2);

  auto cvt = [&](const float* s, __bf16* d, int n) {
    cvt_f32_bf16<<<(n + 255) / 256, 256, 0, stream>>>(s, d, n);
  };
  cvt(hs, hs_b, MROWS * D_MODEL);
  cvt(wq, wq_b, 2048 * 2048);
  cvt(wk, wk_b, 512 * 2048);
  cvt(wv, wv_b, 512 * 2048);
  cvt(wo, wo_b, 2048 * 2048);

  auto gemm = [&](const __bf16* A, const __bf16* W, void* C,
                  int M, int N, int K, int f32) {
    int waves = (M / 16) * (N / 64);
    gemm_bf16_wmma<<<(waves + 7) / 8, 256, 0, stream>>>(A, W, C, M, N, K, f32);
  };
  // QKV projections
  gemm(hs_b, wq_b, q_b, MROWS, 2048, 2048, 0);
  gemm(hs_b, wk_b, k_b, MROWS, 512, 2048, 0);
  gemm(hs_b, wv_b, v_b, MROWS, 512, 2048, 0);

  // RoPE (fp32 math, bf16 storage)
  rope_kernel<<<(BATCH * SEQ * N_HEADS * 64 + 255) / 256, 256, 0, stream>>>(q_b, N_HEADS);
  rope_kernel<<<(BATCH * SEQ * N_KV * 64 + 255) / 256, 256, 0, stream>>>(k_b, N_KV);

  // Sliding-window attention: 4096 waves, 4 waves/block (LDS: 36 KB/WG)
  swa_attn_wmma<<<(BATCH * N_HEADS * (SEQ / 16)) / AWAVES, 128, 0, stream>>>
      (q_b, k_b, v_b, at_b);

  // Output projection, fp32 result
  gemm(at_b, wo_b, (float*)d_out, MROWS, 2048, 2048, 1);
}
